// LlamaMoC_autograd_6579889898128
// MI455X (gfx1250) — compile-verified, matrix operand order checked by reading, not run on hardware
//
#include <hip/hip_runtime.h>
#include <hip/hip_bf16.h>

// ---------------------------------------------------------------------------
// LLaMA MoC MLP for MI455X (gfx1250), wave32 + WMMA bf16 + TDM staging.
//   dims: B=4, S=2048 -> M=8192 tokens; H=2048; I=5632; K_sel=512
// Pipeline:
//   1) cvt fp32 -> bf16 (x, Wg, Wu, Wd)
//   2) fused gate/up GEMM: TDM (tensor_load_to_lds) double-buffered tiles,
//      v_wmma_f32_16x16x32_bf16, epilogue silu(g)*u
//   3) per-token exact top-512 radix-select, zero non-selected sv
//   4) down GEMM (same TDM+WMMA structure) -> fp32 out
// ---------------------------------------------------------------------------

typedef __attribute__((ext_vector_type(16))) __bf16 v16bf;
typedef __attribute__((ext_vector_type(8)))  __bf16 v8bf;
typedef __attribute__((ext_vector_type(8)))  float  v8f;
typedef __attribute__((ext_vector_type(4)))  unsigned int u32x4;
typedef __attribute__((ext_vector_type(8)))  int          i32x8;
typedef __attribute__((ext_vector_type(4)))  int          i32x4;

#define M_TOK 8192
#define H_DIM 2048
#define I_DIM 5632
#define K_SEL 512

// LDS tile geometry: rows of 64 bf16 (128 B) padded by TDM with 16 B -> 144 B
// row stride; 16 consecutive rows then map to 16 distinct LDS banks.
#define TILE_K      64
#define LDS_ROW_B   144

// ---------------------------------------------------------------------------
// Tensor Data Mover: 2-D tile load, global -> LDS.  rows x 64 bf16 elements,
// row stride `stride_elems` in global memory, padded rows in LDS.
static __device__ __forceinline__ void tdm_load_2d(unsigned lds_off,
                                                   const void* gptr,
                                                   int rows, int stride_elems) {
  unsigned long long ga = (unsigned long long)gptr;
  u32x4 g0;
  g0[0] = 1u;                                    // count=1, user descriptor
  g0[1] = lds_off;                               // lds_addr (bytes)
  g0[2] = (unsigned)(ga & 0xFFFFFFFFu);          // global_addr[31:0]
  g0[3] = (unsigned)((ga >> 32) & 0x1FFFFFFu)    // global_addr[56:32]
          | 0x80000000u;                         // type = 2 ("image")

  const unsigned td0 = 0x40000000u;              // huge tensor dims: no OOB
  const unsigned td1 = 0x40000000u;
  i32x8 g1;
  // dword0: data_size=2B (1<<16) | pad_enable (1<<20) |
  //         pad_interval: 32 DW (4<<22) | pad_amount: 4 DW (3<<25)
  g1[0] = (1 << 16) | (1 << 20) | (4 << 22) | (3 << 25);
  g1[1] = (int)((td0 & 0xFFFFu) << 16);                    // tensor_dim0[15:0]
  g1[2] = (int)((td0 >> 16) | ((td1 & 0xFFFFu) << 16));    // td0[31:16]|td1[15:0]
  g1[3] = (int)((td1 >> 16) | (64u << 16));                // td1[31:16]|tile_dim0=64
  g1[4] = (int)(unsigned)rows;                             // tile_dim1 | tile_dim2=0
  g1[5] = (int)(unsigned)stride_elems;                     // tensor_dim0_stride[31:0]
  g1[6] = 0;                                               // stride hi | dim1_stride lo
  g1[7] = 0;
  i32x4 g2 = {0, 0, 0, 0};
  i32x4 g3 = {0, 0, 0, 0};
#if defined(__clang_major__) && (__clang_major__ >= 23)
  i32x8 g4 = {0, 0, 0, 0, 0, 0, 0, 0};
  __builtin_amdgcn_tensor_load_to_lds(g0, g1, g2, g3, g4, 0);
#else
  __builtin_amdgcn_tensor_load_to_lds(g0, g1, g2, g3, 0);
#endif
}

// Load one 16-bit WMMA A/B fragment from a padded LDS tile.
// ISA 16-bit 16x32 layout: lane L -> row L&15, K-chunks at kb=(L>>4)*8:
// elements [kb..kb+7] and [kb+16..kb+23].  Two ds_load_b128 per fragment.
static __device__ __forceinline__ v16bf frag_lds(const char* base, int row,
                                                 int col) {
  const __bf16* p = (const __bf16*)(base + row * LDS_ROW_B + col * 2);
  v8bf lo = *(const v8bf*)(p);
  v8bf hi = *(const v8bf*)(p + 16);
  v16bf r;
#pragma unroll
  for (int i = 0; i < 8; ++i) { r[i] = lo[i]; r[i + 8] = hi[i]; }
  return r;
}

static __device__ __forceinline__ v8f wmma_bf16(v16bf a, v16bf b, v8f c) {
  return __builtin_amdgcn_wmma_f32_16x16x32_bf16(
      false, a, false, b, (short)0, c, false, false);
}

// -------------------------------- stage 1 ----------------------------------
__global__ __launch_bounds__(256) void cvt_f32_bf16(const float* __restrict__ in,
                                                    __bf16* __restrict__ out,
                                                    size_t n) {
  size_t i = (size_t)blockIdx.x * blockDim.x + threadIdx.x;
  size_t stride = (size_t)gridDim.x * blockDim.x;
  for (; i < n; i += stride) out[i] = (__bf16)in[i];
}

// -------------------------------- stage 2 ----------------------------------
// Block tile 128(M) x 64(N), 8 waves as 4(M) x 2(N), wave tile 32x32 for both
// gate and up.  Double-buffered TDM staging: A 128x64 (18432 B padded),
// Bg 64x64 (9216 B), Bu 64x64 (9216 B) per stage => 2 x 36864 B LDS.
#define GU_A_B   (128 * LDS_ROW_B)   // 18432
#define GU_B_B   (64 * LDS_ROW_B)    //  9216
#define GU_ST_B  (GU_A_B + 2 * GU_B_B)

__global__ __launch_bounds__(256) void gateup_gemm(
    const __bf16* __restrict__ xb, const __bf16* __restrict__ wg,
    const __bf16* __restrict__ wu, float* __restrict__ gate,
    __bf16* __restrict__ sv) {
  __shared__ char smem[2 * GU_ST_B];

  const int wid   = threadIdx.x >> 5;
  const int waveM = wid & 3;
  const int waveN = wid >> 2;
  const int m0 = blockIdx.y * 128 + waveM * 32;
  const int n0 = blockIdx.x * 64  + waveN * 32;

  const int lane  = threadIdx.x & 31;
  const int lrow  = lane & 15;
  const int khalf = lane >> 4;
  const int kb    = khalf * 8;

  const unsigned smem_off = (unsigned)(unsigned long long)(void*)smem;
  const __bf16* gA = xb + (size_t)(blockIdx.y * 128) * H_DIM;
  const __bf16* gG = wg + (size_t)(blockIdx.x * 64)  * H_DIM;
  const __bf16* gU = wu + (size_t)(blockIdx.x * 64)  * H_DIM;

  v8f accg[2][2] = {};
  v8f accu[2][2] = {};

  if (wid == 0) {  // prologue: stage 0 <- k=0
    tdm_load_2d(smem_off,                    gA, 128, H_DIM);
    tdm_load_2d(smem_off + GU_A_B,           gG, 64,  H_DIM);
    tdm_load_2d(smem_off + GU_A_B + GU_B_B,  gU, 64,  H_DIM);
  }

  for (int kt = 0; kt < H_DIM / TILE_K; ++kt) {
    const int st = kt & 1;
    if (wid == 0) __builtin_amdgcn_s_wait_tensorcnt(0);
    __syncthreads();  // stage `st` ready; previous compute on st^1 done

    if (wid == 0 && kt + 1 < H_DIM / TILE_K) {  // prefetch stage st^1
      const unsigned so = smem_off + (st ^ 1) * GU_ST_B;
      const int k = (kt + 1) * TILE_K;
      tdm_load_2d(so,                   gA + k, 128, H_DIM);
      tdm_load_2d(so + GU_A_B,          gG + k, 64,  H_DIM);
      tdm_load_2d(so + GU_A_B + GU_B_B, gU + k, 64,  H_DIM);
    }

    const char* sA = smem + st * GU_ST_B;
    const char* sG = sA + GU_A_B;
    const char* sU = sG + GU_B_B;

#pragma unroll
    for (int ks = 0; ks < 2; ++ks) {
      const int col = ks * 32 + kb;
      v16bf a[2], bg[2], bu[2];
#pragma unroll
      for (int mi = 0; mi < 2; ++mi)
        a[mi] = frag_lds(sA, waveM * 32 + mi * 16 + lrow, col);
#pragma unroll
      for (int ni = 0; ni < 2; ++ni) {
        bg[ni] = frag_lds(sG, waveN * 32 + ni * 16 + lrow, col);
        bu[ni] = frag_lds(sU, waveN * 32 + ni * 16 + lrow, col);
      }
#pragma unroll
      for (int mi = 0; mi < 2; ++mi)
#pragma unroll
        for (int ni = 0; ni < 2; ++ni) {
          accg[mi][ni] = wmma_bf16(a[mi], bg[ni], accg[mi][ni]);
          accu[mi][ni] = wmma_bf16(a[mi], bu[ni], accu[mi][ni]);
        }
    }
  }

  // Epilogue: C/D layout -> row = m0+mi*16+(lane>=16?8:0)+e, col = n0+ni*16+lrow
#pragma unroll
  for (int mi = 0; mi < 2; ++mi)
#pragma unroll
    for (int ni = 0; ni < 2; ++ni) {
      const int col = n0 + ni * 16 + lrow;
#pragma unroll
      for (int e = 0; e < 8; ++e) {
        const int row = m0 + mi * 16 + khalf * 8 + e;
        const float g = accg[mi][ni][e];
        const float u = accu[mi][ni][e];
        const float s = g / (1.0f + __expf(-g)) * u;  // silu(g)*u
        const size_t off = (size_t)row * I_DIM + col;
        gate[off] = g;
        sv[off]   = (__bf16)s;
      }
    }
}

// -------------------------------- stage 3 ----------------------------------
// One 256-thread block per token; exact top-K via 4x8-bit radix select on
// monotonic uint keys in LDS; zero sv entries not selected.
__global__ __launch_bounds__(256) void topk_mask(const float* __restrict__ gate,
                                                 __bf16* __restrict__ sv) {
  __shared__ unsigned keys[I_DIM];
  __shared__ unsigned hist[256];
  __shared__ unsigned bc[2];
  __shared__ unsigned eqctr;

  const int row = blockIdx.x;
  const int tid = threadIdx.x;
  const float* grow = gate + (size_t)row * I_DIM;

  for (int i = tid; i < I_DIM; i += 256) {
    unsigned u = __float_as_uint(grow[i]);
    keys[i] = (u & 0x80000000u) ? ~u : (u | 0x80000000u);
  }

  unsigned prefix = 0, prefmask = 0, kth = K_SEL;
#pragma unroll
  for (int shift = 24; shift >= 0; shift -= 8) {
    hist[tid] = 0;
    __syncthreads();
    for (int i = tid; i < I_DIM; i += 256) {
      const unsigned key = keys[i];
      if ((key & prefmask) == prefix) atomicAdd(&hist[(key >> shift) & 0xFFu], 1u);
    }
    __syncthreads();
    if (tid == 0) {
      unsigned running = 0;
      int d = 255;
      for (;; --d) {
        const unsigned c = hist[d];
        if (running + c >= kth || d == 0) { bc[0] = (unsigned)d; bc[1] = kth - running; break; }
        running += c;
      }
    }
    __syncthreads();
    prefix   |= bc[0] << shift;
    prefmask |= 0xFFu << shift;
    kth = bc[1];
    __syncthreads();
  }
  if (tid == 0) eqctr = 0;
  __syncthreads();
  for (int i = tid; i < I_DIM; i += 256) {
    const unsigned key = keys[i];
    bool keep;
    if (key > prefix)       keep = true;
    else if (key == prefix) keep = (atomicAdd(&eqctr, 1u) < kth);
    else                    keep = false;
    if (!keep) sv[(size_t)row * I_DIM + i] = (__bf16)0.0f;
  }
}

// -------------------------------- stage 4 ----------------------------------
// out[m,h] = sum_i sv[m,i] * Wd[h,i]; same TDM-staged structure, K = I_DIM.
#define DN_A_B   (128 * LDS_ROW_B)
#define DN_B_B   (64 * LDS_ROW_B)
#define DN_ST_B  (DN_A_B + DN_B_B)

__global__ __launch_bounds__(256) void down_gemm(const __bf16* __restrict__ sv,
                                                 const __bf16* __restrict__ wd,
                                                 float* __restrict__ out) {
  __shared__ char smem[2 * DN_ST_B];

  const int wid   = threadIdx.x >> 5;
  const int waveM = wid & 3;
  const int waveN = wid >> 2;
  const int m0 = blockIdx.y * 128 + waveM * 32;
  const int n0 = blockIdx.x * 64  + waveN * 32;

  const int lane  = threadIdx.x & 31;
  const int lrow  = lane & 15;
  const int khalf = lane >> 4;
  const int kb    = khalf * 8;

  const unsigned smem_off = (unsigned)(unsigned long long)(void*)smem;
  const __bf16* gA = sv + (size_t)(blockIdx.y * 128) * I_DIM;
  const __bf16* gB = wd + (size_t)(blockIdx.x * 64)  * I_DIM;

  v8f acc[2][2] = {};

  if (wid == 0) {
    tdm_load_2d(smem_off,          gA, 128, I_DIM);
    tdm_load_2d(smem_off + DN_A_B, gB, 64,  I_DIM);
  }

  for (int kt = 0; kt < I_DIM / TILE_K; ++kt) {
    const int st = kt & 1;
    if (wid == 0) __builtin_amdgcn_s_wait_tensorcnt(0);
    __syncthreads();

    if (wid == 0 && kt + 1 < I_DIM / TILE_K) {
      const unsigned so = smem_off + (st ^ 1) * DN_ST_B;
      const int k = (kt + 1) * TILE_K;
      tdm_load_2d(so,          gA + k, 128, I_DIM);
      tdm_load_2d(so + DN_A_B, gB + k, 64,  I_DIM);
    }

    const char* sA = smem + st * DN_ST_B;
    const char* sB = sA + DN_A_B;

#pragma unroll
    for (int ks = 0; ks < 2; ++ks) {
      const int col = ks * 32 + kb;
      v16bf a[2], b[2];
#pragma unroll
      for (int mi = 0; mi < 2; ++mi)
        a[mi] = frag_lds(sA, waveM * 32 + mi * 16 + lrow, col);
#pragma unroll
      for (int ni = 0; ni < 2; ++ni)
        b[ni] = frag_lds(sB, waveN * 32 + ni * 16 + lrow, col);
#pragma unroll
      for (int mi = 0; mi < 2; ++mi)
#pragma unroll
        for (int ni = 0; ni < 2; ++ni)
          acc[mi][ni] = wmma_bf16(a[mi], b[ni], acc[mi][ni]);
    }
  }

#pragma unroll
  for (int mi = 0; mi < 2; ++mi)
#pragma unroll
    for (int ni = 0; ni < 2; ++ni) {
      const int col = n0 + ni * 16 + lrow;
#pragma unroll
      for (int e = 0; e < 8; ++e) {
        const int row = m0 + mi * 16 + khalf * 8 + e;
        out[(size_t)row * H_DIM + col] = acc[mi][ni][e];
      }
    }
}

// ---------------------------------------------------------------------------
extern "C" void kernel_launch(void* const* d_in, const int* in_sizes, int n_in,
                              void* d_out, int out_size, void* d_ws, size_t ws_size,
                              hipStream_t stream) {
  (void)in_sizes; (void)n_in; (void)out_size; (void)ws_size;

  const float* x  = (const float*)d_in[0];  // [8192, 2048]
  const float* Wg = (const float*)d_in[1];  // [5632, 2048]
  const float* Wu = (const float*)d_in[2];  // [5632, 2048]
  const float* Wd = (const float*)d_in[3];  // [2048, 5632]
  float* out = (float*)d_out;               // [8192, 2048]

  const size_t xN = (size_t)M_TOK * H_DIM;
  const size_t wN = (size_t)I_DIM * H_DIM;
  const size_t gN = (size_t)M_TOK * I_DIM;

  char* ws = (char*)d_ws;
  __bf16* xb   = (__bf16*)(ws);
  __bf16* wgb  = (__bf16*)(ws + xN * 2);
  __bf16* wub  = (__bf16*)(ws + xN * 2 + wN * 2);
  __bf16* wdb  = (__bf16*)(ws + xN * 2 + wN * 4);
  float*  gate = (float*) (ws + xN * 2 + wN * 6);
  __bf16* sv   = (__bf16*)(ws + xN * 2 + wN * 6 + gN * 4);

  cvt_f32_bf16<<<4096, 256, 0, stream>>>(x,  xb,  xN);
  cvt_f32_bf16<<<4096, 256, 0, stream>>>(Wg, wgb, wN);
  cvt_f32_bf16<<<4096, 256, 0, stream>>>(Wu, wub, wN);
  cvt_f32_bf16<<<4096, 256, 0, stream>>>(Wd, wdb, wN);

  gateup_gemm<<<dim3(I_DIM / 64, M_TOK / 128), 256, 0, stream>>>(xb, wgb, wub,
                                                                 gate, sv);
  topk_mask<<<M_TOK, 256, 0, stream>>>(gate, sv);

  down_gemm<<<dim3(H_DIM / 64, M_TOK / 128), 256, 0, stream>>>(sv, wdb, out);
}
